// SelfAttention_22557168239120
// MI455X (gfx1250) — matrix operand order repaired and checked by reading, not compiled, not run on hardware
//
#include <hip/hip_runtime.h>
#include <hip/hip_bf16.h>

typedef __attribute__((ext_vector_type(16))) _Float16 v16h;
typedef __attribute__((ext_vector_type(8)))  _Float16 v8h;
typedef __attribute__((ext_vector_type(4)))  _Float16 v4h;
typedef __attribute__((ext_vector_type(8)))  float    v8f;

#define DIM   1024
#define BATCH 4
#define SEQ   2048
#define BS    (BATCH * SEQ)        // 8192 rows total
#define LDSK  40                   // 32 K-halves + 8 pad (80B row stride)
#define STG   136                  // staging tile row stride in halves (128 + 8 pad)
// shared buffer: max(2 K-tiles of 128*LDSK halves, one 128x136 staging tile)
#define SMEM_HALVES (128 * STG)    // 17408 > 2*128*40 = 10240

static __device__ inline v8f zero8() {
    v8f v;
#pragma unroll
    for (int i = 0; i < 8; ++i) v[i] = 0.0f;
    return v;
}

// A fragment (16x32 f16, MxK): lane half selects K blocks {0-7,16-23} / {8-15,24-31}
static __device__ inline v16h load_frag_a(const _Float16* As, int row, int lane) {
    const int half = lane >> 4;
    const _Float16* p = As + row * LDSK + half * 8;
    union { v16h v; v8h h[2]; } u;
    u.h[0] = *(const v8h*)(p);        // K = half*8 + 0..7
    u.h[1] = *(const v8h*)(p + 16);   // K = 16 + half*8 + 0..7
    return u.v;
}

// B fragment (32x16 f16, KxN) stored as N x K rows in LDS: lane half selects K blocks {0-7,16-23} / {8-15,24-31}
static __device__ inline v16h load_frag_b(const _Float16* Bs, int col, int lane) {
    const int half = lane >> 4;
    const _Float16* p = Bs + col * LDSK + half * 8;
    union { v16h v; v8h h[2]; } u;
    u.h[0] = *(const v8h*)(p);
    u.h[1] = *(const v8h*)(p + 16);
    return u.v;
}

// Generic f16 WMMA GEMM:  C[m,n] = scale * sum_k A[m,k]*B[n,k] + bias[n]
// A is MxK row-major (lda). If !TRANSB, B is NxK row-major (ldb) — i.e. C = A·Bᵀ.
// If TRANSB, B source is KxN row-major (ldb) and is transposed during the LDS store phase.
// Block tile 128x128, K step 32, 128 threads = 4 waves (2 M x 2 N), wave tile 64x64.
// Software pipelined: global loads for tile k+1 issue before the WMMAs for tile k.
// OUT_F16: write f16 via LDS-staged coalesced b128 stores. Else write f32 direct.
template <bool TRANSB, bool OUT_F16, bool HAS_BIAS>
__global__ __launch_bounds__(128) void gemm_wmma_f16(
    const _Float16* __restrict__ A, const _Float16* __restrict__ B,
    float* __restrict__ Cf, _Float16* __restrict__ Ch,
    const float* __restrict__ bias,
    int K, int lda, int ldb, int ldc,
    long long strideA, long long strideB, long long strideC,
    float scale, int causal)
{
    __shared__ _Float16 smem[SMEM_HALVES];
    _Float16* As = smem;
    _Float16* Bs = smem + 128 * LDSK;

    const int n0 = blockIdx.x * 128;
    const int m0 = blockIdx.y * 128;
    if (causal && n0 > m0 + 127) return;   // tile entirely above the causal diagonal

    const long long bz = blockIdx.z;
    A += bz * strideA;
    B += bz * strideB;
    const long long cbase = bz * strideC;

    const int tid  = threadIdx.x;
    const int lane = tid & 31;
    const int wid  = tid >> 5;
    const int mW   = (wid & 1) * 64;   // wave M origin in tile
    const int nW   = (wid >> 1) * 64;  // wave N origin in tile
    const int l15  = lane & 15;

    v8h ra[4], rb[4];   // global->LDS staging registers (keeps 8 loads in flight)

    // issue all 8 global_load_b128 for tile at kk0, then (uniform) prefetch kk0+32
    auto load_tiles = [&](int kk0, bool pf) {
#pragma unroll
        for (int it = 0; it < 4; ++it) {
            const int chunk = tid + it * 128;
            const int r = chunk >> 2;
            const int c = (chunk & 3) * 8;
            ra[it] = *(const v8h*)(A + (long long)(m0 + r) * lda + kk0 + c);
        }
#pragma unroll
        for (int it = 0; it < 4; ++it) {
            const int chunk = tid + it * 128;
            if (!TRANSB) {
                const int r = chunk >> 2;
                const int c = (chunk & 3) * 8;
                rb[it] = *(const v8h*)(B + (long long)(n0 + r) * ldb + kk0 + c);
            } else {
                const int kk = chunk >> 4;
                const int nn = (chunk & 15) * 8;
                rb[it] = *(const v8h*)(B + (long long)(kk0 + kk) * ldb + n0 + nn);
            }
        }
        if (pf) {   // uniform branch; prefetch the tile after this one
#pragma unroll
            for (int it = 0; it < 4; ++it) {
                const int chunk = tid + it * 128;
                const int r = chunk >> 2;
                const int c = (chunk & 3) * 8;
                __builtin_prefetch(A + (long long)(m0 + r) * lda + kk0 + 32 + c, 0, 1);
                if (!TRANSB)
                    __builtin_prefetch(B + (long long)(n0 + r) * ldb + kk0 + 32 + c, 0, 1);
                else
                    __builtin_prefetch(B + (long long)(kk0 + 32 + (chunk >> 4)) * ldb +
                                       n0 + (chunk & 15) * 8, 0, 1);
            }
        }
    };

    auto store_tiles = [&]() {
#pragma unroll
        for (int it = 0; it < 4; ++it) {
            const int chunk = tid + it * 128;
            const int r = chunk >> 2;
            const int c = (chunk & 3) * 8;
            *(v8h*)&As[r * LDSK + c] = ra[it];
        }
#pragma unroll
        for (int it = 0; it < 4; ++it) {
            const int chunk = tid + it * 128;
            if (!TRANSB) {
                const int r = chunk >> 2;
                const int c = (chunk & 3) * 8;
                *(v8h*)&Bs[r * LDSK + c] = rb[it];
            } else {
                const int kk = chunk >> 4;
                const int nn = (chunk & 15) * 8;
#pragma unroll
                for (int j = 0; j < 8; ++j) Bs[(nn + j) * LDSK + kk] = rb[it][j];
            }
        }
    };

    v8f acc[4][4];
#pragma unroll
    for (int i = 0; i < 4; ++i)
#pragma unroll
        for (int j = 0; j < 4; ++j) acc[i][j] = zero8();

    load_tiles(0, true);
    for (int k0 = 0; k0 < K; k0 += 32) {
        store_tiles();
        __syncthreads();
        if (k0 + 32 < K) load_tiles(k0 + 32, k0 + 64 < K);  // overlap with WMMAs below

        v16h afrag[4], bfrag[4];
#pragma unroll
        for (int i = 0; i < 4; ++i) afrag[i] = load_frag_a(As, mW + i * 16 + l15, lane);
#pragma unroll
        for (int j = 0; j < 4; ++j) bfrag[j] = load_frag_b(Bs, nW + j * 16 + l15, lane);
#pragma unroll
        for (int i = 0; i < 4; ++i)
#pragma unroll
            for (int j = 0; j < 4; ++j)
                acc[i][j] = __builtin_amdgcn_wmma_f32_16x16x32_f16(
                    false, afrag[i], false, bfrag[j],
                    (short)0, acc[i][j], false, false);
        __syncthreads();
    }

    // ---- epilogue: C frag layout row = (lane>>4)*8 + v, col = lane&15 ----
    float bvals[4];
#pragma unroll
    for (int j = 0; j < 4; ++j)
        bvals[j] = HAS_BIAS ? bias[n0 + nW + j * 16 + l15] : 0.0f;

    if (OUT_F16) {
        // stage f16 tile in LDS (reuses As/Bs space; safe after final barrier)
#pragma unroll
        for (int i = 0; i < 4; ++i) {
            const int ml = mW + i * 16 + (lane >> 4) * 8;
#pragma unroll
            for (int j = 0; j < 4; ++j) {
                const int nl = nW + j * 16 + l15;
#pragma unroll
                for (int v = 0; v < 8; ++v)
                    smem[(ml + v) * STG + nl] =
                        (_Float16)(acc[i][j][v] * scale + bvals[j]);
            }
        }
        __syncthreads();
        // coalesced b128 stores: 2048 chunks of 8 halves, 16 per thread
#pragma unroll
        for (int it = 0; it < 16; ++it) {
            const int chunk = tid + it * 128;
            const int r = chunk >> 4;
            const int c = (chunk & 15) * 8;
            _Float16* g = Ch + cbase + (long long)(m0 + r) * ldc + n0 + c;
            *(v8h*)g = *(const v8h*)&smem[r * STG + c];
        }
    } else {
#pragma unroll
        for (int i = 0; i < 4; ++i) {
            const int mbase = m0 + mW + i * 16 + (lane >> 4) * 8;
#pragma unroll
            for (int j = 0; j < 4; ++j) {
                const long long base =
                    cbase + (long long)mbase * ldc + (n0 + nW + j * 16 + l15);
#pragma unroll
                for (int v = 0; v < 8; ++v)
                    Cf[base + (long long)v * ldc] = acc[i][j][v] * scale + bvals[j];
            }
        }
    }
}

// fp32 -> f16 convert, 4 elements/thread (n must be a multiple of 4)
__global__ __launch_bounds__(256) void cvt_f32_f16(
    const float* __restrict__ src, _Float16* __restrict__ dst, int n)
{
    const int i = (blockIdx.x * 256 + threadIdx.x) * 4;
    if (i < n) {
        float4 f = *(const float4*)(src + i);
        v4h h;
        h[0] = (_Float16)f.x; h[1] = (_Float16)f.y;
        h[2] = (_Float16)f.z; h[3] = (_Float16)f.w;
        *(v4h*)(dst + i) = h;
    }
}

// Causal softmax over one score row; writes normalized f16 P *in place* over the
// f32 row (P row r lives in the first 2*SEQ bytes of score row r), zero-fills j>i.
__global__ __launch_bounds__(256) void softmax_causal(
    const float* __restrict__ Sc, _Float16* __restrict__ P)
{
    __shared__ float row[SEQ];
    __shared__ float red[256];
    const int i = blockIdx.x, b = blockIdx.y, tid = threadIdx.x;
    const long long r = (long long)b * SEQ + i;
    const float* sp = Sc + r * SEQ;
    _Float16*  pp = P + r * (2 * SEQ);   // f16 elements; aliases sp's bytes (safe per-row)
    const int len = i + 1;

    float lmax = -3.4e38f;
    for (int j = tid; j < len; j += 256) {
        const float v = sp[j];
        row[j] = v;
        lmax = fmaxf(lmax, v);
    }
    red[tid] = lmax;
    __syncthreads();
    for (int s = 128; s > 0; s >>= 1) {
        if (tid < s) red[tid] = fmaxf(red[tid], red[tid + s]);
        __syncthreads();
    }
    const float m = red[0];
    __syncthreads();

    float lsum = 0.0f;
    for (int j = tid; j < len; j += 256) {
        const float e = __expf(row[j] - m);
        row[j] = e;
        lsum += e;
    }
    red[tid] = lsum;
    __syncthreads();
    for (int s = 128; s > 0; s >>= 1) {
        if (tid < s) red[tid] += red[tid + s];
        __syncthreads();
    }
    const float inv = 1.0f / red[0];
    __syncthreads();

    for (int j = tid; j < len; j += 256) pp[j] = (_Float16)(row[j] * inv);
    for (int j = len + tid; j < SEQ; j += 256) pp[j] = (_Float16)0.0f;
}

extern "C" void kernel_launch(void* const* d_in, const int* in_sizes, int n_in,
                              void* d_out, int out_size, void* d_ws, size_t ws_size,
                              hipStream_t stream) {
    (void)in_sizes; (void)n_in; (void)out_size; (void)ws_size;
    const float* X  = (const float*)d_in[0];
    // d_in[1] = attention_mask: causal tril, implemented directly — never read
    const float* Wq = (const float*)d_in[2];
    const float* bq = (const float*)d_in[3];
    const float* Wk = (const float*)d_in[4];
    const float* bk = (const float*)d_in[5];
    const float* Wv = (const float*)d_in[6];
    const float* bv = (const float*)d_in[7];
    const float* Wo = (const float*)d_in[8];
    const float* bo = (const float*)d_in[9];
    float* out = (float*)d_out;

    // ---- workspace layout (~152 MB) ----
    char* w = (char*)d_ws;
    _Float16* Xh   = (_Float16*)w; w += (size_t)BS * DIM * 2;        // 16 MB
    _Float16* Wcat = (_Float16*)w; w += (size_t)3 * DIM * DIM * 2;   //  6 MB [Wq;Wk;Wv]
    _Float16* Woh  = (_Float16*)w; w += (size_t)DIM * DIM * 2;       //  2 MB
    float*    bcat = (float*)w;    w += (size_t)3 * DIM * 4;         // 12 KB
    _Float16* QKV  = (_Float16*)w; w += (size_t)BS * 3 * DIM * 2;    // 48 MB
    float*    Sc   = (float*)w;    w += (size_t)BATCH * SEQ * SEQ * 4; // 64 MB (reused by P)
    _Float16* Oh   = (_Float16*)w;                                   // 16 MB
    _Float16* P    = (_Float16*)Sc;  // in-place f16 overlay, row stride 2*SEQ

    // ---- 1. fp32 -> f16 converts + weight packing ----
    cvt_f32_f16<<<(BS * DIM) / 1024, 256, 0, stream>>>(X, Xh, BS * DIM);
    cvt_f32_f16<<<(DIM * DIM) / 1024, 256, 0, stream>>>(Wq, Wcat, DIM * DIM);
    cvt_f32_f16<<<(DIM * DIM) / 1024, 256, 0, stream>>>(Wk, Wcat + DIM * DIM, DIM * DIM);
    cvt_f32_f16<<<(DIM * DIM) / 1024, 256, 0, stream>>>(Wv, Wcat + 2 * DIM * DIM, DIM * DIM);
    cvt_f32_f16<<<(DIM * DIM) / 1024, 256, 0, stream>>>(Wo, Woh, DIM * DIM);
    hipMemcpyAsync(bcat,            bq, DIM * 4, hipMemcpyDeviceToDevice, stream);
    hipMemcpyAsync(bcat + DIM,      bk, DIM * 4, hipMemcpyDeviceToDevice, stream);
    hipMemcpyAsync(bcat + 2 * DIM,  bv, DIM * 4, hipMemcpyDeviceToDevice, stream);

    // ---- 2. fused QKV projection: (8192x1024)·(3072x1024)ᵀ + bias -> f16 ----
    gemm_wmma_f16<false, true, true><<<dim3(3 * DIM / 128, BS / 128, 1), 128, 0, stream>>>(
        Xh, Wcat, nullptr, QKV, bcat,
        DIM, DIM, DIM, 3 * DIM, 0LL, 0LL, 0LL, 1.0f, 0);

    // ---- 3. scores = (Q·Kᵀ)/32 per batch, causal tiles only -> f32 ----
    gemm_wmma_f16<false, false, false><<<dim3(SEQ / 128, SEQ / 128, BATCH), 128, 0, stream>>>(
        QKV /*Q cols 0..1023*/, QKV + DIM /*K cols*/, Sc, nullptr, nullptr,
        DIM, 3 * DIM, 3 * DIM, SEQ,
        (long long)SEQ * 3 * DIM, (long long)SEQ * 3 * DIM, (long long)SEQ * SEQ,
        0.03125f, 1);

    // ---- 4. causal softmax -> f16 P (in-place over score rows) ----
    softmax_causal<<<dim3(SEQ, BATCH), 256, 0, stream>>>(Sc, P);

    // ---- 5. O = P·V  (V is KxN -> transposed into LDS) -> f16 ----
    gemm_wmma_f16<true, true, false><<<dim3(DIM / 128, SEQ / 128, BATCH), 128, 0, stream>>>(
        P, QKV + 2 * DIM /*V cols*/, nullptr, Oh, nullptr,
        SEQ, 2 * SEQ, 3 * DIM, DIM,
        (long long)2 * SEQ * SEQ, (long long)SEQ * 3 * DIM, (long long)SEQ * DIM,
        1.0f, 0);

    // ---- 6. out = O·Woᵀ + bo -> fp32 ----
    gemm_wmma_f16<false, false, true><<<dim3(DIM / 128, BS / 128, 1), 128, 0, stream>>>(
        Oh, Woh, out, nullptr, bo,
        DIM, DIM, DIM, DIM, 0LL, 0LL, 0LL, 1.0f, 0);
}
